// AttentionMILModel_57561151701574
// MI455X (gfx1250) — matrix operand order repaired and checked by reading, not compiled, not run on hardware
//
#include <hip/hip_runtime.h>
#include <hip/hip_bf16.h>

typedef __attribute__((ext_vector_type(16))) __bf16 v16bf;
typedef __attribute__((ext_vector_type(8)))  __bf16 v8bf;
typedef __attribute__((ext_vector_type(8)))  float  v8f;

#define DD 256      // feature dim
#define AA 128      // attention hidden dim
#define NBAG 64

// ---- order-preserving float <-> uint encoding for atomic max ----
__device__ __forceinline__ unsigned fenc(float f) {
    unsigned u = __float_as_uint(f);
    return (u & 0x80000000u) ? ~u : (u | 0x80000000u);
}
__device__ __forceinline__ float fdec(unsigned e) {
    unsigned u = (e & 0x80000000u) ? (e ^ 0x80000000u) : ~e;
    return __uint_as_float(u);
}

// ---------------------------------------------------------------------------
// Kernel 0: pre-swizzle W1 [256][128] fp32 -> bf16 WMMA B-fragments.
// Fragment order: [kb(8)][nt(8)][lane(32)][e(16)]; lane: col = nt*16+(lane&15),
// element e holds K = kb*32 + 16*(lane>>4) + e   (contiguous k order, same as the A fragment).
// ---------------------------------------------------------------------------
__global__ void prep_kernel(const float* __restrict__ W1, __bf16* __restrict__ W1frag) {
    int idx = blockIdx.x * blockDim.x + threadIdx.x;   // 2048 fragments
    if (idx >= 8 * 8 * 32) return;
    int kb   = idx >> 8;
    int nt   = (idx >> 5) & 7;
    int lane = idx & 31;
    int hi   = lane >> 4;
    int ncol = nt * 16 + (lane & 15);
    __bf16* dst = W1frag + (size_t)idx * 16;
#pragma unroll
    for (int e = 0; e < 16; ++e) {
        int k = kb * 32 + 16 * hi + e;
        dst[e] = (__bf16)W1[k * AA + ncol];
    }
}

__global__ void init_kernel(unsigned* bmax, float* denom, float* pooled) {
    int t = blockIdx.x * blockDim.x + threadIdx.x;
    if (t < NBAG) { bmax[t] = 0u; denom[t] = 0.0f; }
    for (int i = t; i < NBAG * DD; i += gridDim.x * blockDim.x) pooled[i] = 0.0f;
}

// ---------------------------------------------------------------------------
// Kernel 1: scores[i] = w2 . tanh(x[i,:] @ W1 + b1) + b2   via bf16 WMMA.
// Block = 128 threads (4 waves), 64 rows per block, 16 rows per wave.
// ---------------------------------------------------------------------------
__global__ __launch_bounds__(128) void scores_kernel(
    const float* __restrict__ x, const __bf16* __restrict__ W1frag,
    const float* __restrict__ b1, const float* __restrict__ w2,
    const float* __restrict__ b2p, float* __restrict__ scores) {

    // padded row stride: 264 bf16 = 528 B (16B-aligned, avoids 64-bank stride)
    __shared__ __bf16 xb[64][264];

    const int tid = threadIdx.x;
    const size_t rowbase = (size_t)blockIdx.x * 64;

    // cooperative load: 64x256 fp32 tile -> bf16 in LDS (coalesced float4)
    {
        const float4* xg = (const float4*)(x + rowbase * DD);
        for (int i = tid; i < 64 * DD / 4; i += 128) {
            float4 v = xg[i];
            int r = i >> 6;            // 64 float4 per row
            int c = (i & 63) * 4;
            xb[r][c + 0] = (__bf16)v.x;
            xb[r][c + 1] = (__bf16)v.y;
            xb[r][c + 2] = (__bf16)v.z;
            xb[r][c + 3] = (__bf16)v.w;
        }
    }
    __syncthreads();

    const int wave = tid >> 5;
    const int lane = tid & 31;
    const int m    = lane & 15;
    const int hi   = lane >> 4;
    const int mrow = wave * 16 + m;    // A-matrix row for this lane

    v8f cacc[8];
#pragma unroll
    for (int nt = 0; nt < 8; ++nt) cacc[nt] = {};

#pragma unroll
    for (int kb = 0; kb < 8; ++kb) {
        // A fragment: e0..15 -> K=kb*32+16*hi+e (matches W1frag B order)
        const v8bf* p0 = (const v8bf*)&xb[mrow][kb * 32 + 16 * hi];
        v8bf alo = p0[0];
        v8bf ahi = p0[1];   // +8 elements
        v16bf afrag = __builtin_shufflevector(alo, ahi,
            0, 1, 2, 3, 4, 5, 6, 7, 8, 9, 10, 11, 12, 13, 14, 15);
#pragma unroll
        for (int nt = 0; nt < 8; ++nt) {
            v16bf bfrag = *(const v16bf*)(W1frag + ((size_t)((kb * 8 + nt) * 32 + lane)) * 16);
            cacc[nt] = __builtin_amdgcn_wmma_f32_16x16x32_bf16(
                false, afrag, false, bfrag, (short)0, cacc[nt], false, false);
        }
    }

    // epilogue: tanh + dot with w2; C layout: lane covers col n = nt*16+m,
    // rows (within 16-row tile) = 8*hi + j for j=0..7
    float accj[8];
#pragma unroll
    for (int j = 0; j < 8; ++j) accj[j] = 0.0f;
#pragma unroll
    for (int nt = 0; nt < 8; ++nt) {
        int n = nt * 16 + m;
        float b1n = b1[n];
        float w2n = w2[n];
#pragma unroll
        for (int j = 0; j < 8; ++j)
            accj[j] += tanhf(cacc[nt][j] + b1n) * w2n;
    }
    // reduce over the 16 lanes sharing the same rows
#pragma unroll
    for (int mask = 8; mask >= 1; mask >>= 1) {
#pragma unroll
        for (int j = 0; j < 8; ++j)
            accj[j] += __shfl_xor(accj[j], mask, 16);
    }
    if (m == 0) {
        float bb = *b2p;
        size_t r0 = rowbase + (size_t)wave * 16 + 8 * hi;
#pragma unroll
        for (int j = 0; j < 8; ++j) scores[r0 + j] = accj[j] + bb;
    }
}

// ---------------------------------------------------------------------------
// Kernel 2: per-bag max of scores (encoded atomicMax, wave-uniform fast path)
// ---------------------------------------------------------------------------
__global__ void bagmax_kernel(const int* __restrict__ seg,
                              const float* __restrict__ scores,
                              unsigned* __restrict__ bmax, int N) {
    int i = blockIdx.x * blockDim.x + threadIdx.x;
    if (i >= N) return;
    int bag = seg[i];
    unsigned e = fenc(scores[i]);
    int bag0 = __shfl(bag, 0, 32);
    if (__all(bag == bag0)) {
#pragma unroll
        for (int msk = 16; msk >= 1; msk >>= 1) {
            unsigned o = __shfl_xor(e, msk, 32);
            e = (o > e) ? o : e;
        }
        if ((threadIdx.x & 31) == 0) atomicMax(&bmax[bag0], e);
    } else {
        atomicMax(&bmax[bag], e);
    }
}

// ---------------------------------------------------------------------------
// Kernel 3: per-bag sum of exp(score - max)
// ---------------------------------------------------------------------------
__global__ void denom_kernel(const int* __restrict__ seg,
                             const float* __restrict__ scores,
                             const unsigned* __restrict__ bmax,
                             float* __restrict__ denom, int N) {
    int i = blockIdx.x * blockDim.x + threadIdx.x;
    if (i >= N) return;
    int bag = seg[i];
    float ev = __expf(scores[i] - fdec(bmax[bag]));
    int bag0 = __shfl(bag, 0, 32);
    if (__all(bag == bag0)) {
#pragma unroll
        for (int msk = 16; msk >= 1; msk >>= 1)
            ev += __shfl_xor(ev, msk, 32);
        if ((threadIdx.x & 31) == 0) atomicAdd(&denom[bag0], ev);
    } else {
        atomicAdd(&denom[bag], ev);
    }
}

// ---------------------------------------------------------------------------
// Kernel 4: pooled[bag,:] += a_i * x[i,:]. Thread = one column; register
// accumulation across rows, flush on bag transition (segments are sorted).
// ---------------------------------------------------------------------------
__global__ __launch_bounds__(256) void pool_kernel(
    const float* __restrict__ x, const int* __restrict__ seg,
    const float* __restrict__ scores, const unsigned* __restrict__ bmax,
    const float* __restrict__ denom, float* __restrict__ pooled) {

    __shared__ float a_s[256];
    __shared__ int   seg_s[256];

    const int t = threadIdx.x;
    const size_t r0 = (size_t)blockIdx.x * 256;

    {
        size_t r = r0 + t;
        int bag = seg[r];
        float av = __expf(scores[r] - fdec(bmax[bag])) / denom[bag];
        a_s[t] = av;
        seg_s[t] = bag;
    }
    __syncthreads();

    float acc = 0.0f;
    int cur = seg_s[0];
    for (int i = 0; i < 256; ++i) {
        int b = seg_s[i];
        if (b != cur) {
            atomicAdd(&pooled[(size_t)cur * DD + t], acc);
            acc = 0.0f;
            cur = b;
        }
        acc += x[(r0 + i) * DD + t] * a_s[i];
    }
    atomicAdd(&pooled[(size_t)cur * DD + t], acc);
}

// ---------------------------------------------------------------------------
// Kernel 5: out[64][2] = pooled @ Wh + bh
// ---------------------------------------------------------------------------
__global__ void head_kernel(const float* __restrict__ pooled,
                            const float* __restrict__ Wh,
                            const float* __restrict__ bh,
                            float* __restrict__ out) {
    int t = threadIdx.x;          // 128 = 64 bags * 2 outputs
    int b = t >> 1, o = t & 1;
    float s = bh[o];
    for (int d = 0; d < DD; ++d) s += pooled[b * DD + d] * Wh[d * 2 + o];
    out[t] = s;
}

extern "C" void kernel_launch(void* const* d_in, const int* in_sizes, int n_in,
                              void* d_out, int out_size, void* d_ws, size_t ws_size,
                              hipStream_t stream) {
    const float* x   = (const float*)d_in[0];
    const int*   seg = (const int*)d_in[1];
    const float* W1  = (const float*)d_in[3];
    const float* b1  = (const float*)d_in[4];
    const float* w2  = (const float*)d_in[5];
    const float* b2  = (const float*)d_in[6];
    const float* Wh  = (const float*)d_in[7];
    const float* bh  = (const float*)d_in[8];
    float* out = (float*)d_out;

    const int N = in_sizes[0] / DD;   // 524288

    // workspace layout
    char* ws = (char*)d_ws;
    __bf16*   W1frag = (__bf16*)ws;                               // 64 KB
    float*    scores = (float*)(ws + 65536);                      // N*4 B
    unsigned* bmax   = (unsigned*)(ws + 65536 + (size_t)N * 4);   // 256 B
    float*    denom  = (float*)(ws + 65536 + (size_t)N * 4 + 256);
    float*    pooled = (float*)(ws + 65536 + (size_t)N * 4 + 512);// 64 KB

    prep_kernel  <<<8,      256, 0, stream>>>(W1, W1frag);
    init_kernel  <<<64,     256, 0, stream>>>(bmax, denom, pooled);
    scores_kernel<<<N / 64, 128, 0, stream>>>(x, W1frag, b1, w2, b2, scores);
    bagmax_kernel<<<N / 256, 256, 0, stream>>>(seg, scores, bmax, N);
    denom_kernel <<<N / 256, 256, 0, stream>>>(seg, scores, bmax, denom, N);
    pool_kernel  <<<N / 256, 256, 0, stream>>>(x, seg, scores, bmax, denom, pooled);
    head_kernel  <<<1,      128, 0, stream>>>(pooled, Wh, bh, out);
}
